// BeamSearchDecoder_85341000172151
// MI455X (gfx1250) — matrix operand order repaired and checked by reading, not compiled, not run on hardware
//
#include <hip/hip_runtime.h>
#include <hip/hip_bf16.h>

#define PAD_ID 0
#define BOS_ID 101
#define Dm   768
#define Hn   12
#define DHn  64
#define DFFn 3072
#define Vsz  30522

typedef __attribute__((ext_vector_type(16))) __bf16 v16bf;
typedef __attribute__((ext_vector_type(8)))  float  v8f;

typedef float  f32x4  __attribute__((ext_vector_type(4)));
typedef f32x4  f32x4u __attribute__((aligned(4)));     // under-aligned view (4B)
typedef __bf16 bf16x8 __attribute__((ext_vector_type(8)));
typedef bf16x8 bf16x8u __attribute__((aligned(4)));

__device__ __forceinline__ float warp_sum32(float v) {
#pragma unroll
  for (int o = 16; o > 0; o >>= 1) v += __shfl_xor(v, o, 32);
  return v;
}
__device__ __forceinline__ float warp_max32(float v) {
#pragma unroll
  for (int o = 16; o > 0; o >>= 1) v = fmaxf(v, __shfl_xor(v, o, 32));
  return v;
}

__device__ __forceinline__ __bf16 to_bf16(float v)  { return (__bf16)v; }
__device__ __forceinline__ __bf16 to_bf16(__bf16 v) { return v; }

// A fragment: two contiguous 8-float runs -> 4x global_load_b128 + pk cvt.
// Frag e in 0..7 -> K=kb+e ; e in 8..15 -> K=kb+16+(e-8)  (p already at +kb)
__device__ __forceinline__ void load_a_frag(const float* p, v16bf& af) {
  f32x4 a0 = *(const f32x4u*)(p);
  f32x4 a1 = *(const f32x4u*)(p + 4);
  f32x4 a2 = *(const f32x4u*)(p + 16);
  f32x4 a3 = *(const f32x4u*)(p + 20);
#pragma unroll
  for (int e = 0; e < 4; ++e) {
    af[e]      = (__bf16)a0[e];
    af[4 + e]  = (__bf16)a1[e];
    af[8 + e]  = (__bf16)a2[e];
    af[12 + e] = (__bf16)a3[e];
  }
}

// Vectorized B-fragment loaders (16 contiguous elements at q).
__device__ __forceinline__ void load_b_vec(const float* q, v16bf& bf) {
  const f32x4u* p = (const f32x4u*)q;
  f32x4 b0 = p[0], b1 = p[1], b2 = p[2], b3 = p[3];
#pragma unroll
  for (int e = 0; e < 4; ++e) {
    bf[e]      = (__bf16)b0[e];
    bf[4 + e]  = (__bf16)b1[e];
    bf[8 + e]  = (__bf16)b2[e];
    bf[12 + e] = (__bf16)b3[e];
  }
}
__device__ __forceinline__ void load_b_vec(const __bf16* q, v16bf& bf) {
  const bf16x8u* p = (const bf16x8u*)q;
  bf16x8 b0 = p[0], b1 = p[1];
#pragma unroll
  for (int e = 0; e < 8; ++e) { bf[e] = b0[e]; bf[8 + e] = b1[e]; }
}

// ---------------------------------------------------------------------------
// Strided/batched GEMM on the CDNA5 WMMA path. One wave computes a 16x64 C
// tile (NT=4 sub-tiles): one A fragment feeds 4 back-to-back v_wmma ops.
// Constraints by construction: sAk==1, sCn==1, Kd % 32 == 0, grid covers N.
// Edge handling is clamp-only (no value selects needed): A row m / B column n
// only contribute to C row m / column n, and OOB rows/cols are never stored.
// The full/edge decision is hoisted out of the K-loop so the hot loop is
// spill-free straight-line code.
// ---------------------------------------------------------------------------
template <typename TB, int NT, bool VB>
__global__ void wmma_gemm_kernel(
    const float* __restrict__ A, const TB* __restrict__ Bm, float* __restrict__ C,
    int M, int N, int Kd,
    long long sAm, long long sBk, long long sBn, long long sCm,
    int innerB,
    long long aBO, long long aBI, long long bBO, long long bBI,
    long long cBO, long long cBI)
{
  const int lane = threadIdx.x & 31;
  const int tn0 = blockIdx.x * (16 * NT);
  const int tm  = blockIdx.y * 16;
  const int bo  = blockIdx.z / innerB, bi = blockIdx.z % innerB;
  const float* Ab = A  + (long long)bo * aBO + (long long)bi * aBI;
  const TB*    Bb = Bm + (long long)bo * bBO + (long long)bi * bBI;
  float*       Cb = C  + (long long)bo * cBO + (long long)bi * cBI;

  const int m  = tm + (lane & 15);
  const int mc = (m < M) ? m : (M - 1);            // clamp only
  const int kb = (lane >> 4) << 3;                 // 0 or 8
  const float* pA = Ab + (long long)mc * sAm + kb;

  v8f acc[NT];
#pragma unroll
  for (int t = 0; t < NT; ++t) acc[t] = (v8f){0.f, 0.f, 0.f, 0.f, 0.f, 0.f, 0.f, 0.f};

  const long long bStep = (long long)32 * sBk;

  if (VB) {
    const TB* pB = Bb + (long long)kb * sBk + tn0;   // lane%16 = N column, elems = K (table order)
    if (tn0 + 16 * NT <= N) {
      // ---------------- full tile: pure vector loads ----------------
      for (int k0 = 0; k0 < Kd; k0 += 32) {
        v16bf afrag;
        load_a_frag(pA, afrag);
        pA += 32;
#pragma unroll
        for (int t = 0; t < NT; ++t) {
          v16bf bfrag;
#pragma unroll
          for (int e = 0; e < 16; ++e)
            bfrag[e] = to_bf16(pB[(long long)(e + ((e >> 3) << 3)) * sBk + 16 * t + (lane & 15)]);
          acc[t] = __builtin_amdgcn_wmma_f32_16x16x32_bf16(
              false, afrag, false, bfrag, (short)0, acc[t], false, false);
        }
        pB += bStep;
      }
    } else {
      // ---------------- N-edge tile: clamped scalar loads ----------------
      for (int k0 = 0; k0 < Kd; k0 += 32) {
        v16bf afrag;
        load_a_frag(pA, afrag);
        pA += 32;
#pragma unroll
        for (int t = 0; t < NT; ++t) {
          TB raw[16];
#pragma unroll
          for (int e = 0; e < 16; ++e) {
            const int n = tn0 + 16 * t + (lane & 15);
            const int off = ((n < N) ? n : (N - 1)) - tn0;   // >= 0
            raw[e] = pB[(long long)(e + ((e >> 3) << 3)) * sBk + off];
          }
          v16bf bfrag;
#pragma unroll
          for (int e = 0; e < 16; ++e) bfrag[e] = to_bf16(raw[e]);
          acc[t] = __builtin_amdgcn_wmma_f32_16x16x32_bf16(
              false, afrag, false, bfrag, (short)0, acc[t], false, false);
        }
        pB += bStep;
      }
    }
  } else {
    // ---------------- strided B (attention K^T) ----------------
    const TB* pB = Bb + (long long)kb * sBk;
    for (int k0 = 0; k0 < Kd; k0 += 32) {
      v16bf afrag;
      load_a_frag(pA, afrag);
      pA += 32;
#pragma unroll
      for (int t = 0; t < NT; ++t) {
        TB raw[16];
#pragma unroll
        for (int e = 0; e < 16; ++e) {
          const int n = tn0 + 16 * t + (lane & 15);
          const int nc = (n < N) ? n : (N - 1);
          raw[e] = pB[(long long)(e + ((e >> 3) << 3)) * sBk + (long long)nc * sBn];
        }
        v16bf bfrag;
#pragma unroll
        for (int e = 0; e < 16; ++e) bfrag[e] = to_bf16(raw[e]);
        acc[t] = __builtin_amdgcn_wmma_f32_16x16x32_bf16(
            false, afrag, false, bfrag, (short)0, acc[t], false, false);
      }
      pB += bStep;
    }
  }

  // ---- store: lane%16 = column, 8 rows per lane (coalesced b32 stores) ----
  const int nl    = lane & 15;
  const int mBase = tm + ((lane >> 4) << 3);
#pragma unroll
  for (int t = 0; t < NT; ++t) {
    const int nOut = tn0 + 16 * t + nl;
    if (nOut < N) {
#pragma unroll
      for (int r = 0; r < 8; ++r) {
        const int mm = mBase + r;
        if (mm < M) Cb[(long long)mm * sCm + nOut] = acc[t][r];
      }
    }
  }
}

__global__ void f32_to_bf16_kernel(const float* __restrict__ in, __bf16* __restrict__ out,
                                   long long n) {
  long long i = (long long)blockIdx.x * blockDim.x + threadIdx.x;
  if (i < n) out[i] = (__bf16)in[i];
}

// h[b,s,:] = emb[X[b,s]] + pos[s]; mask[b,s] = (X!=PAD) ? 0 : -1e9. Wave/row.
__global__ void embed_enc_kernel(const int* __restrict__ X, const float* __restrict__ emb,
                                 const float* __restrict__ pos, float* __restrict__ h,
                                 float* __restrict__ mask, int S) {
  const int row = blockIdx.x;            // b*S + s
  const int s = row % S;
  const int tok = X[row];
  const float* e = emb + (long long)tok * Dm;
  const float* p = pos + (long long)s * Dm;
  float* out = h + (long long)row * Dm;
  for (int d = threadIdx.x; d < Dm; d += 32) out[d] = e[d] + p[d];
  if (threadIdx.x == 0) mask[row] = (tok != PAD_ID) ? 0.0f : -1e9f;
}

// x[b,k,:] = emb[tokens[b,k]] + pos[t]. Wave per (b,k).
__global__ void embed_dec_kernel(const int* __restrict__ tokens, const float* __restrict__ emb,
                                 const float* __restrict__ pos, int t, float* __restrict__ x) {
  const int row = blockIdx.x;            // b*K + k
  const int tok = tokens[row];
  const float* e = emb + (long long)tok * Dm;
  const float* p = pos + (long long)t * Dm;
  float* out = x + (long long)row * Dm;
  for (int d = threadIdx.x; d < Dm; d += 32) out[d] = e[d] + p[d];
}

// att rows of length S: softmax(att*scale + mask[b,:]) in place. Wave per row.
__global__ void softmax_mask_kernel(float* __restrict__ att, const float* __restrict__ mask,
                                    int Mrows, int S, float scale) {
  const long long row = blockIdx.x;          // (b*H + h)*Mrows + i
  const long long bh = row / Mrows;
  const int b = (int)(bh / Hn);
  float* a = att + row * (long long)S;
  const float* mb = mask + (long long)b * S;
  const int lane = threadIdx.x;

  float lmax = -1e30f;
  for (int j = lane; j < S; j += 32) {
    float v = a[j] * scale + mb[j];
    a[j] = v;
    lmax = fmaxf(lmax, v);
  }
  lmax = warp_max32(lmax);
  float lsum = 0.f;
  for (int j = lane; j < S; j += 32) {
    float v = __expf(a[j] - lmax);
    a[j] = v;
    lsum += v;
  }
  lsum = warp_sum32(lsum);
  const float inv = 1.0f / lsum;
  for (int j = lane; j < S; j += 32) a[j] *= inv;
}

// out = LayerNorm(a + b), no affine, eps=1e-5. Wave per row of 768 (24/lane).
__global__ void add_ln_kernel(const float* __restrict__ a, const float* __restrict__ b,
                              float* __restrict__ out) {
  const long long row = blockIdx.x;
  const float* pa = a + row * Dm;
  const float* pb = b + row * Dm;
  float* po = out + row * Dm;
  const int lane = threadIdx.x;
  float vals[Dm / 32];
  float s = 0.f, s2 = 0.f;
#pragma unroll
  for (int i = 0; i < Dm / 32; ++i) {
    const int d = lane + 32 * i;
    const float v = pa[d] + pb[d];
    vals[i] = v; s += v; s2 += v * v;
  }
  s = warp_sum32(s); s2 = warp_sum32(s2);
  const float mean = s * (1.0f / Dm);
  const float var = s2 * (1.0f / Dm) - mean * mean;
  const float rstd = rsqrtf(var + 1e-5f);
#pragma unroll
  for (int i = 0; i < Dm / 32; ++i) {
    const int d = lane + 32 * i;
    po[d] = (vals[i] - mean) * rstd;
  }
}

// tanh-approx GELU in place (jax.nn.gelu default).
__global__ void gelu_kernel(float* __restrict__ x, long long n) {
  long long i = (long long)blockIdx.x * blockDim.x + threadIdx.x;
  if (i < n) {
    const float v = x[i];
    const float c = 0.7978845608028654f * (v + 0.044715f * v * v * v);
    x[i] = 0.5f * v * (1.0f + tanhf(c));
  }
}

// Per-row max and log(sum(exp(x-max))) over V columns. Block(256) per row.
__global__ void row_lse_kernel(const float* __restrict__ logits, float* __restrict__ rmax,
                               float* __restrict__ rlse, int cols) {
  const int row = blockIdx.x;
  const float* p = logits + (long long)row * cols;
  __shared__ float red[256];
  const int tid = threadIdx.x;
  float m = -1e30f;
  for (int j = tid; j < cols; j += 256) m = fmaxf(m, p[j]);
  red[tid] = m; __syncthreads();
  for (int s = 128; s > 0; s >>= 1) {
    if (tid < s) red[tid] = fmaxf(red[tid], red[tid + s]);
    __syncthreads();
  }
  m = red[0]; __syncthreads();
  float sum = 0.f;
  for (int j = tid; j < cols; j += 256) sum += __expf(p[j] - m);
  red[tid] = sum; __syncthreads();
  for (int s = 128; s > 0; s >>= 1) {
    if (tid < s) red[tid] += red[tid + s];
    __syncthreads();
  }
  if (tid == 0) { rmax[row] = m; rlse[row] = logf(red[0]); }
}

// Per-batch top-K over K*V of scores[k] + log_softmax(logits[k,:]).
// Updates scores/tokens in place, records back-pointers + tokens for step t.
__global__ void topk_kernel(const float* __restrict__ logits, const float* __restrict__ rmax,
                            const float* __restrict__ rlse, float* __restrict__ scores,
                            int* __restrict__ tokens, int* __restrict__ beam_hist_t,
                            int* __restrict__ tok_hist_t, int K) {
  const int b = blockIdx.x;
  const int tid = threadIdx.x;
  __shared__ float cvals[256 * 4];
  __shared__ int   cidx[256 * 4];

  float off[4];
#pragma unroll
  for (int k = 0; k < 4; ++k)
    off[k] = scores[b * K + k] - rmax[b * K + k] - rlse[b * K + k];

  const long long total = (long long)K * Vsz;
  float best[4] = {-1e30f, -1e30f, -1e30f, -1e30f};
  int   bix[4]  = {0, 0, 0, 0};
  for (long long i = tid; i < total; i += 256) {
    const int k = (int)(i / Vsz);
    const float v = off[k] + logits[(long long)(b * K + k) * Vsz + (i - (long long)k * Vsz)];
    if (v > best[3]) {
      best[3] = v; bix[3] = (int)i;
#pragma unroll
      for (int p = 3; p > 0; --p) {
        if (best[p] > best[p - 1]) {
          float tv = best[p]; best[p] = best[p - 1]; best[p - 1] = tv;
          int ti = bix[p]; bix[p] = bix[p - 1]; bix[p - 1] = ti;
        }
      }
    }
  }
#pragma unroll
  for (int k = 0; k < 4; ++k) { cvals[tid * 4 + k] = best[k]; cidx[tid * 4 + k] = bix[k]; }
  __syncthreads();

  if (tid == 0) {
    for (int sel = 0; sel < K; ++sel) {
      float bm = -1e30f; int bj = 0;
      for (int j = 0; j < 256 * 4; ++j)
        if (cvals[j] > bm) { bm = cvals[j]; bj = j; }
      const int gi = cidx[bj];
      cvals[bj] = -1e31f;
      const int beam = gi / Vsz;
      const int tok  = gi % Vsz;
      scores[b * K + sel]      = bm;
      tokens[b * K + sel]      = tok;
      beam_hist_t[b * K + sel] = beam;
      tok_hist_t[b * K + sel]  = tok;
    }
  }
}

__global__ void init_beam_kernel(int* __restrict__ tokens, float* __restrict__ scores, int K) {
  const int i = threadIdx.x;             // B*K = 32 <= blockDim
  tokens[i] = BOS_ID;
  scores[i] = ((i % K) == 0) ? 0.0f : -1e9f;
}

// Backtrack through back-pointers; write seq[B,K,T] (int32) and scores (f32).
__global__ void backtrack_kernel(const int* __restrict__ beam_hist, const int* __restrict__ tok_hist,
                                 const float* __restrict__ scores, int* __restrict__ seq_out,
                                 float* __restrict__ scores_out, int T, int B, int K) {
  const int i = threadIdx.x;             // b*K + k
  if (i >= B * K) return;
  const int b = i / K, k = i % K;
  int ptr = k;
  for (int t = T - 1; t >= 0; --t) {
    const int tok = tok_hist[(long long)t * B * K + b * K + ptr];
    seq_out[(long long)(b * K + k) * T + t] = tok;
    ptr = beam_hist[(long long)t * B * K + b * K + ptr];
  }
  scores_out[i] = scores[i];
}

// ---------------------------------------------------------------------------
template <typename TB, bool VB>
static void gemm(hipStream_t st, const float* A, const TB* Bm, float* C,
                 int M, int N, int Kd,
                 long long sAm, long long sBk, long long sBn, long long sCm,
                 int batches = 1, int innerB = 1,
                 long long aBO = 0, long long aBI = 0,
                 long long bBO = 0, long long bBI = 0,
                 long long cBO = 0, long long cBI = 0) {
  dim3 g((N + 63) / 64, (M + 15) / 16, batches);
  wmma_gemm_kernel<TB, 4, VB><<<g, 32, 0, st>>>(A, Bm, C, M, N, Kd, sAm, sBk, sBn,
                                                sCm, innerB, aBO, aBI, bBO, bBI, cBO, cBI);
}

extern "C" void kernel_launch(void* const* d_in, const int* in_sizes, int n_in,
                              void* d_out, int out_size, void* d_ws, size_t ws_size,
                              hipStream_t stream) {
  (void)in_sizes; (void)n_in; (void)out_size; (void)ws_size;
  const int B = 8, S = 128, K = 4, T = 32;
  const float scale = 0.125f;           // 1/sqrt(64)

  const int*   X    = (const int*)d_in[0];
  const float* emb  = (const float*)d_in[1];
  const float* pos  = (const float*)d_in[2];
  const float* Wq   = (const float*)d_in[3];
  const float* Wk   = (const float*)d_in[4];
  const float* Wv   = (const float*)d_in[5];
  const float* Wo   = (const float*)d_in[6];
  const float* W1   = (const float*)d_in[7];
  const float* W2   = (const float*)d_in[8];
  const float* Wdq  = (const float*)d_in[9];
  const float* Wdk  = (const float*)d_in[10];
  const float* Wdv  = (const float*)d_in[11];
  const float* Wdo  = (const float*)d_in[12];
  const float* Wd1  = (const float*)d_in[13];
  const float* Wd2  = (const float*)d_in[14];
  const float* Wvoc = (const float*)d_in[15];

  // ---- workspace bump allocator (256B aligned) ----
  char* wp = (char*)d_ws;
  auto alloc = [&](size_t bytes) -> void* {
    void* p = (void*)wp;
    wp += (bytes + 255) & ~(size_t)255;
    return p;
  };
  const int BS = B * S;                  // 1024 encoder rows
  const int BK = B * K;                  // 32 decoder rows

  float* e_h    = (float*)alloc((size_t)BS * Dm * 4);
  float* mask   = (float*)alloc((size_t)BS * 4);
  float* e_q    = (float*)alloc((size_t)BS * Dm * 4);
  float* e_k    = (float*)alloc((size_t)BS * Dm * 4);
  float* e_v    = (float*)alloc((size_t)BS * Dm * 4);
  float* e_att  = (float*)alloc((size_t)B * Hn * S * S * 4);
  float* e_ctx  = (float*)alloc((size_t)BS * Dm * 4);
  float* e_ao   = (float*)alloc((size_t)BS * Dm * 4);
  float* e_h1   = (float*)alloc((size_t)BS * Dm * 4);
  float* e_ffn  = (float*)alloc((size_t)BS * DFFn * 4);
  float* e_ffn2 = (float*)alloc((size_t)BS * Dm * 4);
  float* e_mem  = (float*)alloc((size_t)BS * Dm * 4);
  float* Kmem   = (float*)alloc((size_t)BS * Dm * 4);
  float* Vmem   = (float*)alloc((size_t)BS * Dm * 4);

  float* d_x    = (float*)alloc((size_t)BK * Dm * 4);
  float* d_qd   = (float*)alloc((size_t)BK * Dm * 4);
  float* d_att  = (float*)alloc((size_t)B * Hn * K * S * 4);
  float* d_c    = (float*)alloc((size_t)BK * Dm * 4);
  float* d_c2   = (float*)alloc((size_t)BK * Dm * 4);
  float* d_h1   = (float*)alloc((size_t)BK * Dm * 4);
  float* d_ffn  = (float*)alloc((size_t)BK * DFFn * 4);
  float* d_ffn2 = (float*)alloc((size_t)BK * Dm * 4);
  float* d_h2   = (float*)alloc((size_t)BK * Dm * 4);
  float* logits = (float*)alloc((size_t)BK * Vsz * 4);
  float* rmax   = (float*)alloc((size_t)BK * 4);
  float* rlse   = (float*)alloc((size_t)BK * 4);
  float* scoresB= (float*)alloc((size_t)BK * 4);
  int*   tokens = (int*)  alloc((size_t)BK * 4);
  int*   beamH  = (int*)  alloc((size_t)T * BK * 4);
  int*   tokH   = (int*)  alloc((size_t)T * BK * 4);
  __bf16* WvocBF= (__bf16*)alloc((size_t)Dm * Vsz * 2 + 64);

  // ---- one-time (per launch): bf16 copy of the dominant weight stream ----
  {
    long long nW = (long long)Dm * Vsz;
    f32_to_bf16_kernel<<<(unsigned)((nW + 255) / 256), 256, 0, stream>>>(Wvoc, WvocBF, nW);
  }

  // =================== encoder ===================
  embed_enc_kernel<<<BS, 32, 0, stream>>>(X, emb, pos, e_h, mask, S);

  gemm<float, true>(stream, e_h, Wq, e_q, BS, Dm, Dm, Dm, Dm, 1, Dm);
  gemm<float, true>(stream, e_h, Wk, e_k, BS, Dm, Dm, Dm, Dm, 1, Dm);
  gemm<float, true>(stream, e_h, Wv, e_v, BS, Dm, Dm, Dm, Dm, 1, Dm);

  // scores[b,h,i,j] = q[b,i,h,:] . k[b,j,h,:]   (batched over b*H; B strided)
  gemm<float, false>(stream, e_q, e_k, e_att, S, S, DHn,
                     /*sAm*/ Dm, /*sBk*/ 1, /*sBn*/ Dm, /*sCm*/ S,
                     B * Hn, Hn,
                     (long long)S * Dm, DHn, (long long)S * Dm, DHn,
                     (long long)Hn * S * S, (long long)S * S);
  softmax_mask_kernel<<<B * Hn * S, 32, 0, stream>>>(e_att, mask, S, S, scale);
  // ctx[b,i,h,:] = att[b,h,i,:] @ v[b,:,h,:]
  gemm<float, true>(stream, e_att, e_v, e_ctx, S, DHn, S,
                    /*sAm*/ S, /*sBk*/ Dm, /*sBn*/ 1, /*sCm*/ Dm,
                    B * Hn, Hn,
                    (long long)Hn * S * S, (long long)S * S, (long long)S * Dm, DHn,
                    (long long)S * Dm, DHn);
  gemm<float, true>(stream, e_ctx, Wo, e_ao, BS, Dm, Dm, Dm, Dm, 1, Dm);
  add_ln_kernel<<<BS, 32, 0, stream>>>(e_h, e_ao, e_h1);

  gemm<float, true>(stream, e_h1, W1, e_ffn, BS, DFFn, Dm, Dm, DFFn, 1, DFFn);
  gelu_kernel<<<(unsigned)(((long long)BS * DFFn + 255) / 256), 256, 0, stream>>>(
      e_ffn, (long long)BS * DFFn);
  gemm<float, true>(stream, e_ffn, W2, e_ffn2, BS, Dm, DFFn, DFFn, Dm, 1, Dm);
  add_ln_kernel<<<BS, 32, 0, stream>>>(e_h1, e_ffn2, e_mem);

  gemm<float, true>(stream, e_mem, Wdk, Kmem, BS, Dm, Dm, Dm, Dm, 1, Dm);
  gemm<float, true>(stream, e_mem, Wdv, Vmem, BS, Dm, Dm, Dm, Dm, 1, Dm);

  // =================== beam-search decode ===================
  init_beam_kernel<<<1, 32, 0, stream>>>(tokens, scoresB, K);

  for (int t = 0; t < T; ++t) {
    embed_dec_kernel<<<BK, 32, 0, stream>>>(tokens, emb, pos, t, d_x);
    gemm<float, true>(stream, d_x, Wdq, d_qd, BK, Dm, Dm, Dm, Dm, 1, Dm);

    // a[b,h,k,s] = qd[b,k,h,:] . Kmem[b,s,h,:]
    gemm<float, false>(stream, d_qd, Kmem, d_att, K, S, DHn,
                       /*sAm*/ Dm, /*sBk*/ 1, /*sBn*/ Dm, /*sCm*/ S,
                       B * Hn, Hn,
                       (long long)K * Dm, DHn, (long long)S * Dm, DHn,
                       (long long)Hn * K * S, (long long)K * S);
    softmax_mask_kernel<<<B * Hn * K, 32, 0, stream>>>(d_att, mask, K, S, scale);
    // c[b,k,h,:] = a[b,h,k,:] @ Vmem[b,:,h,:]
    gemm<float, true>(stream, d_att, Vmem, d_c, K, DHn, S,
                      /*sAm*/ S, /*sBk*/ Dm, /*sBn*/ 1, /*sCm*/ Dm,
                      B * Hn, Hn,
                      (long long)Hn * K * S, (long long)K * S, (long long)S * Dm, DHn,
                      (long long)K * Dm, DHn);
    gemm<float, true>(stream, d_c, Wdo, d_c2, BK, Dm, Dm, Dm, Dm, 1, Dm);
    add_ln_kernel<<<BK, 32, 0, stream>>>(d_x, d_c2, d_h1);

    gemm<float, true>(stream, d_h1, Wd1, d_ffn, BK, DFFn, Dm, Dm, DFFn, 1, DFFn);
    gelu_kernel<<<(unsigned)(((long long)BK * DFFn + 255) / 256), 256, 0, stream>>>(
        d_ffn, (long long)BK * DFFn);
    gemm<float, true>(stream, d_ffn, Wd2, d_ffn2, BK, Dm, DFFn, DFFn, Dm, 1, Dm);
    add_ln_kernel<<<BK, 32, 0, stream>>>(d_h1, d_ffn2, d_h2);

    // vocab projection: bf16 weight stream (L2-resident, ~47MB/step)
    gemm<__bf16, true>(stream, d_h2, WvocBF, logits, BK, Vsz, Dm, Dm, Vsz, 1, Vsz);

    row_lse_kernel<<<BK, 256, 0, stream>>>(logits, rmax, rlse, Vsz);
    topk_kernel<<<B, 256, 0, stream>>>(logits, rmax, rlse, scoresB, tokens,
                                       beamH + t * BK, tokH + t * BK, K);
  }

  // =================== backtrack -> d_out ===================
  int*   seq_out    = (int*)d_out;                    // [B,K,T] int32
  float* scores_out = (float*)d_out + (size_t)BK * T; // [B,K] fp32
  backtrack_kernel<<<1, 32, 0, stream>>>(beamH, tokH, scoresB, seq_out, scores_out, T, B, K);
}
